// TraceGNN_81192061764391
// MI455X (gfx1250) — matrix operand order repaired and checked by reading, not compiled, not run on hardware
//
#include <hip/hip_runtime.h>
#include <hip/hip_bf16.h>

// ---------------------------------------------------------------------------
// TraceGNN on MI455X (gfx1250): 2-layer GATConv + edge MLP.
// Dense projections use v_wmma_f32_16x16x32_bf16 (wave32 WMMA) with
// pre-packed bf16 transposed weights so A and B fragments are pure b128 loads.
// Edge passes are wave-per-edge coalesced with f32 atomics (memory bound part).
// ---------------------------------------------------------------------------

#define N_NODES 100000
#define N_EDGES 1600000
#define D_IN    64
#define D_H     128

typedef __attribute__((ext_vector_type(16))) __bf16 v16bf;
typedef __attribute__((ext_vector_type(8)))  float  v8f;

union FragBf { unsigned int u[8]; v16bf v; };

__device__ __forceinline__ unsigned short f32_to_bf16_bits(float f) {
    return (unsigned short)(__float_as_uint(f) >> 16);
}

__device__ __forceinline__ float gelu_exact(float v) {
    return 0.5f * v * (1.0f + erff(v * 0.70710678118654752f));
}

// float atomic max via sign-split int atomics (bit-pattern monotonic trick)
__device__ __forceinline__ void atomicMaxFloat(float* addr, float val) {
    int iv = __float_as_int(val);
    if (iv >= 0) atomicMax((int*)addr, iv);
    else         atomicMin((unsigned int*)addr, (unsigned int)iv);
}

// ---------------------------------------------------------------------------
// 0) weight prep: Wt[col][k] = bf16(W[k][col])   (f32 [K,128] -> bf16 [128,K])
//    Runs once per launch; makes B fragments contiguous 32B per lane.
// ---------------------------------------------------------------------------
__global__ void pack_weight_bf16_kernel(const float* __restrict__ W,
                                        unsigned short* __restrict__ Wt,
                                        int K) {
    int i = blockIdx.x * blockDim.x + threadIdx.x;   // over 128*K
    if (i >= D_H * K) return;
    int col = i / K, k = i - col * K;
    Wt[i] = f32_to_bf16_bits(W[(size_t)k * D_H + col]);
}

// ---------------------------------------------------------------------------
// 1) gather: h0_bf16[n,d] = bf16(emb[x[n],d])
// ---------------------------------------------------------------------------
__global__ void gather_embed_kernel(const int* __restrict__ x,
                                    const float* __restrict__ emb,
                                    unsigned short* __restrict__ h0bf,
                                    int total) {
    int i = blockIdx.x * blockDim.x + threadIdx.x;
    if (i >= total) return;
    int n = i >> 6, d = i & 63;
    float f = emb[(size_t)x[n] * D_IN + d];
    h0bf[i] = f32_to_bf16_bits(f);
}

// ---------------------------------------------------------------------------
// 2) WMMA GEMM: Y[N,128] = Xbf16[N,K] @ W[K,128], W pre-packed bf16 as [128,K]
//    block = 256 thr = 8 waves; block -> 16 rows x 128 cols; wave -> 16x16 tile
// ---------------------------------------------------------------------------
template <int K>
__global__ void gemm_bf16_wmma_kernel(const unsigned short* __restrict__ Xbf,
                                      const unsigned short* __restrict__ Wt,
                                      float* __restrict__ Y) {
    const int lane = threadIdx.x & 31;
    const int wave = threadIdx.x >> 5;      // column tile 0..7
    const int row0 = blockIdx.x * 16;       // 6250 blocks * 16 = 100000 exactly
    const int col0 = wave * 16;
    const int m    = lane & 15;             // M (for A) / N (for B,C)
    const int half = lane >> 4;             // lane group 0/1

    v8f acc = {};
    #pragma unroll
    for (int kk = 0; kk < K; kk += 32) {
        // ---- A fragment: 16x32 bf16 (lanes 0-15 -> K 0..15, 16-31 -> K 16..31), 2x b128 per lane ----
        FragBf a;
        const unsigned short* xrow = Xbf + (size_t)(row0 + m) * K + kk;
        #pragma unroll
        for (int v = 0; v < 8; ++v) {
            int kb = half * 16 + v * 2;
            a.u[v] = *(const unsigned int*)(xrow + kb);     // packed bf16 pair
        }
        // ---- B fragment: 32x16; lanes 0-15 -> K 0..15, 16-31 -> K 16..31 ----
        // Wt is [col][k] bf16, so lane reads 16 contiguous elements (2x b128).
        FragBf b;
        const unsigned short* wcol = Wt + (size_t)(col0 + m) * K + kk + half * 16;
        #pragma unroll
        for (int v = 0; v < 8; ++v) {
            b.u[v] = *(const unsigned int*)(wcol + 2 * v);  // packed bf16 pair
        }
        acc = __builtin_amdgcn_wmma_f32_16x16x32_bf16(
                  false, a.v, false, b.v, (short)0, acc, false, false);
    }
    // ---- store C: VGPR v -> row (row0 + v + half*8), col (col0 + m) ----
    #pragma unroll
    for (int v = 0; v < 8; ++v) {
        Y[(size_t)(row0 + v + half * 8) * D_H + (col0 + m)] = acc[v];
    }
}

// ---------------------------------------------------------------------------
// 3) per-node attention dots + (re)init of max / sum accumulators
//    one wave32 per node
// ---------------------------------------------------------------------------
__global__ void node_logits_kernel(const float* __restrict__ hlin,
                                   const float* __restrict__ a_src,
                                   const float* __restrict__ a_dst,
                                   float* __restrict__ as_out,
                                   float* __restrict__ ad_out,
                                   float* __restrict__ mx,
                                   float* __restrict__ ssum) {
    int wid  = (blockIdx.x * blockDim.x + threadIdx.x) >> 5;
    int lane = threadIdx.x & 31;
    if (wid >= N_NODES) return;
    const float* h = hlin + (size_t)wid * D_H;
    float ps = 0.f, pd = 0.f;
    #pragma unroll
    for (int k = 0; k < 4; ++k) {
        int c = lane + k * 32;
        float hv = h[c];
        ps += hv * a_src[c];
        pd += hv * a_dst[c];
    }
    #pragma unroll
    for (int off = 16; off; off >>= 1) {
        ps += __shfl_down(ps, off, 32);
        pd += __shfl_down(pd, off, 32);
    }
    if (lane == 0) {
        as_out[wid] = ps;
        ad_out[wid] = pd;
        mx[wid]   = __uint_as_float(0xFF800000u);   // -inf
        ssum[wid] = 0.0f;
    }
}

__global__ void clear_f32_kernel(float* __restrict__ p, int total) {
    int i = blockIdx.x * blockDim.x + threadIdx.x;
    if (i < total) p[i] = 0.0f;
}

// ---------------------------------------------------------------------------
// 4) edge passes over E + N (self-loop) edges
// ---------------------------------------------------------------------------
__device__ __forceinline__ void edge_sd(const int* ei, int i, int& s, int& d) {
    if (i < N_EDGES) { s = ei[i]; d = ei[N_EDGES + i]; }
    else             { s = d = i - N_EDGES; }
}

__global__ void edge_max_kernel(const int* __restrict__ ei,
                                const float* __restrict__ as_in,
                                const float* __restrict__ ad_in,
                                float* __restrict__ mx) {
    int i = blockIdx.x * blockDim.x + threadIdx.x;
    if (i >= N_EDGES + N_NODES) return;
    int s, d; edge_sd(ei, i, s, d);
    float l = as_in[s] + ad_in[d];
    l = (l < 0.0f) ? l * 0.2f : l;          // leaky_relu 0.2
    atomicMaxFloat(&mx[d], l);
}

__global__ void edge_exp_kernel(const int* __restrict__ ei,
                                const float* __restrict__ as_in,
                                const float* __restrict__ ad_in,
                                const float* __restrict__ mx,
                                float* __restrict__ eedge,
                                float* __restrict__ ssum) {
    int i = blockIdx.x * blockDim.x + threadIdx.x;
    if (i >= N_EDGES + N_NODES) return;
    int s, d; edge_sd(ei, i, s, d);
    float l = as_in[s] + ad_in[d];
    l = (l < 0.0f) ? l * 0.2f : l;
    float e = __expf(l - mx[d]);
    eedge[i] = e;
    atomicAdd(&ssum[d], e);
}

// one wave32 per edge: agg[dst] += alpha * hlin[src]
__global__ void edge_agg_kernel(const int* __restrict__ ei,
                                const float* __restrict__ hlin,
                                const float* __restrict__ eedge,
                                const float* __restrict__ ssum,
                                float* __restrict__ agg) {
    int wid  = (blockIdx.x * blockDim.x + threadIdx.x) >> 5;
    int lane = threadIdx.x & 31;
    if (wid >= N_EDGES + N_NODES) return;
    int s, d; edge_sd(ei, wid, s, d);
    float alpha = eedge[wid] / (ssum[d] + 1e-16f);
    const float* hs = hlin + (size_t)s * D_H;
    float*       od = agg  + (size_t)d * D_H;
    #pragma unroll
    for (int k = 0; k < 4; ++k) {
        int c = lane + k * 32;
        atomicAdd(&od[c], alpha * hs[c]);
    }
}

// ---------------------------------------------------------------------------
// 5) bias + exact GELU; emit bf16 (next GEMM input) and optionally f32
// ---------------------------------------------------------------------------
__global__ void bias_gelu_kernel(const float* __restrict__ agg,
                                 const float* __restrict__ bias,
                                 unsigned short* __restrict__ out_bf,
                                 float* __restrict__ out_f32,
                                 int total) {
    int i = blockIdx.x * blockDim.x + threadIdx.x;
    if (i >= total) return;
    float g = gelu_exact(agg[i] + bias[i & (D_H - 1)]);
    out_bf[i] = f32_to_bf16_bits(g);
    if (out_f32) out_f32[i] = g;
}

// ---------------------------------------------------------------------------
// 6) final edge MLP: gelu([h[src], h[dst]] @ W_fc + b_fc), wave32 per edge
// ---------------------------------------------------------------------------
__global__ void edge_fc_kernel(const int* __restrict__ ei,
                               const float* __restrict__ h,
                               const float* __restrict__ Wfc,
                               const float* __restrict__ bfc,
                               float* __restrict__ out) {
    int wid  = (blockIdx.x * blockDim.x + threadIdx.x) >> 5;
    int lane = threadIdx.x & 31;
    if (wid >= N_EDGES) return;
    int s = ei[wid], d = ei[N_EDGES + wid];
    const float* hs = h + (size_t)s * D_H;
    const float* hd = h + (size_t)d * D_H;
    float p = 0.f;
    #pragma unroll
    for (int k = 0; k < 4; ++k) {
        int c = lane + k * 32;
        p += hs[c] * Wfc[c] + hd[c] * Wfc[D_H + c];
    }
    #pragma unroll
    for (int off = 16; off; off >>= 1) p += __shfl_down(p, off, 32);
    if (lane == 0) out[wid] = gelu_exact(p + bfc[0]);
}

// ---------------------------------------------------------------------------
// launch
// ---------------------------------------------------------------------------
extern "C" void kernel_launch(void* const* d_in, const int* in_sizes, int n_in,
                              void* d_out, int out_size, void* d_ws, size_t ws_size,
                              hipStream_t stream) {
    const int*   x     = (const int*)  d_in[0];
    const int*   ei    = (const int*)  d_in[1];
    const float* emb   = (const float*)d_in[2];
    const float* W1    = (const float*)d_in[3];
    const float* as1   = (const float*)d_in[4];
    const float* ad1   = (const float*)d_in[5];
    const float* b1    = (const float*)d_in[6];
    const float* W2    = (const float*)d_in[7];
    const float* as2   = (const float*)d_in[8];
    const float* ad2   = (const float*)d_in[9];
    const float* b2    = (const float*)d_in[10];
    const float* Wfc   = (const float*)d_in[11];
    const float* bfc   = (const float*)d_in[12];
    float*       out   = (float*)d_out;

    // workspace layout (256B aligned slices)
    char* ws = (char*)d_ws;
    size_t off = 0;
    auto take = [&](size_t bytes) { char* p = ws + off; off += (bytes + 255) & ~(size_t)255; return p; };
    unsigned short* h0bf   = (unsigned short*)take((size_t)N_NODES * D_IN * 2);
    float*          hlin   = (float*)take((size_t)N_NODES * D_H * 4);
    float*          agg    = (float*)take((size_t)N_NODES * D_H * 4);
    unsigned short* hactbf = (unsigned short*)take((size_t)N_NODES * D_H * 2);
    float*          hactf  = (float*)take((size_t)N_NODES * D_H * 4);
    float*          nas    = (float*)take((size_t)N_NODES * 4);
    float*          nad    = (float*)take((size_t)N_NODES * 4);
    float*          nmx    = (float*)take((size_t)N_NODES * 4);
    float*          nsum   = (float*)take((size_t)N_NODES * 4);
    float*          eedge  = (float*)take((size_t)(N_EDGES + N_NODES) * 4);
    unsigned short* Wt1    = (unsigned short*)take((size_t)D_H * D_IN * 2);
    unsigned short* Wt2    = (unsigned short*)take((size_t)D_H * D_H * 2);
    (void)ws_size; (void)n_in; (void)in_sizes; (void)out_size;

    const int TB = 256;
    const int EN = N_EDGES + N_NODES;
    dim3 blk(TB);
    dim3 grid_w1((D_H * D_IN + TB - 1) / TB);
    dim3 grid_w2((D_H * D_H + TB - 1) / TB);
    dim3 grid_gather(((size_t)N_NODES * D_IN + TB - 1) / TB);
    dim3 grid_gemm(N_NODES / 16);                         // 6250
    dim3 grid_nodewave(((size_t)N_NODES * 32 + TB - 1) / TB);
    dim3 grid_nfeat(((size_t)N_NODES * D_H + TB - 1) / TB);
    dim3 grid_edge((EN + TB - 1) / TB);
    dim3 grid_edgewave(((size_t)EN * 32 + TB - 1) / TB);
    dim3 grid_fcwave(((size_t)N_EDGES * 32 + TB - 1) / TB);

    // ---- prep: pack/transpose weights to bf16, gather embeddings ----
    pack_weight_bf16_kernel<<<grid_w1, blk, 0, stream>>>(W1, Wt1, D_IN);
    pack_weight_bf16_kernel<<<grid_w2, blk, 0, stream>>>(W2, Wt2, D_H);
    gather_embed_kernel<<<grid_gather, blk, 0, stream>>>(x, emb, h0bf, N_NODES * D_IN);

    // ---- layer 1 ----
    gemm_bf16_wmma_kernel<D_IN><<<grid_gemm, blk, 0, stream>>>(h0bf, Wt1, hlin);
    node_logits_kernel<<<grid_nodewave, blk, 0, stream>>>(hlin, as1, ad1, nas, nad, nmx, nsum);
    clear_f32_kernel<<<grid_nfeat, blk, 0, stream>>>(agg, N_NODES * D_H);
    edge_max_kernel<<<grid_edge, blk, 0, stream>>>(ei, nas, nad, nmx);
    edge_exp_kernel<<<grid_edge, blk, 0, stream>>>(ei, nas, nad, nmx, eedge, nsum);
    edge_agg_kernel<<<grid_edgewave, blk, 0, stream>>>(ei, hlin, eedge, nsum, agg);
    bias_gelu_kernel<<<grid_nfeat, blk, 0, stream>>>(agg, b1, hactbf, (float*)nullptr, N_NODES * D_H);

    // ---- layer 2 ----
    gemm_bf16_wmma_kernel<D_H><<<grid_gemm, blk, 0, stream>>>(hactbf, Wt2, hlin);
    node_logits_kernel<<<grid_nodewave, blk, 0, stream>>>(hlin, as2, ad2, nas, nad, nmx, nsum);
    clear_f32_kernel<<<grid_nfeat, blk, 0, stream>>>(agg, N_NODES * D_H);
    edge_max_kernel<<<grid_edge, blk, 0, stream>>>(ei, nas, nad, nmx);
    edge_exp_kernel<<<grid_edge, blk, 0, stream>>>(ei, nas, nad, nmx, eedge, nsum);
    edge_agg_kernel<<<grid_edgewave, blk, 0, stream>>>(ei, hlin, eedge, nsum, agg);
    bias_gelu_kernel<<<grid_nfeat, blk, 0, stream>>>(agg, b2, hactbf, hactf, N_NODES * D_H);

    // ---- edge MLP head ----
    edge_fc_kernel<<<grid_fcwave, blk, 0, stream>>>(ei, hactf, Wfc, bfc, out);
}
